// PredictiveDecoder_14285061226781
// MI455X (gfx1250) — matrix operand order repaired and checked by reading, not compiled, not run on hardware
//
#include <hip/hip_runtime.h>
#include <math.h>

// Problem constants (match reference)
#define B_    128
#define P_    625
#define ENC_  512
#define A_    256
#define E_    256
#define H_    512
#define V_    70
#define T_    70
#define START_TOK_ 68

typedef __attribute__((ext_vector_type(16))) __bf16          v16bf;
typedef __attribute__((ext_vector_type(16))) unsigned short  v16u;
typedef __attribute__((ext_vector_type(8)))  unsigned short  v8u;
typedef __attribute__((ext_vector_type(8)))  float           v8f;

__device__ __forceinline__ unsigned short f2bf(float f) {
  unsigned int u = __builtin_bit_cast(unsigned int, f);
  u += 0x7FFFu + ((u >> 16) & 1u);           // round-to-nearest-even
  return (unsigned short)(u >> 16);
}
__device__ __forceinline__ float bf2f(unsigned short s) {
  unsigned int u = ((unsigned int)s) << 16;
  return __builtin_bit_cast(float, u);
}
__device__ __forceinline__ float sigm(float x) { return 1.0f / (1.0f + __expf(-x)); }

__device__ __forceinline__ void store_out(float* p, float v)          { *p = v; }
__device__ __forceinline__ void store_out(unsigned short* p, float v) { *p = f2bf(v); }

__device__ __forceinline__ v16bf wmma_cat(v8u lo, v8u hi) {
  v16u a = __builtin_shufflevector(lo, hi, 0,1,2,3,4,5,6,7,8,9,10,11,12,13,14,15);
  return __builtin_bit_cast(v16bf, a);
}
__device__ __forceinline__ v8f wmma_bf16(v16bf a, v16bf b, v8f c) {
  return __builtin_amdgcn_wmma_f32_16x16x32_bf16(false, a, false, b, (short)0, c, false, false);
}

// ---------------------------------------------------------------------------
// bf16 WMMA GEMM with pre-swizzled B and 32x32 register tile per wave:
//   C[M,N] = A[M,K](bf16 row-major, lda) @ Bs(fragment-major) + bias
// Block = 256 threads = 8 waves arranged 2(M) x 4(N); block tile 64(M) x 128(N).
// Wave tile = 32x32 = 2x2 of 16x16 fragments (4 accumulators) ->
//   4 WMMA per K-chunk off 2 A-frags + 2 B-frags (2 b128 loads per WMMA).
// A-frag (16-bit 16x32 layout): lane<16 -> K {0..7,16..23}; lane>=16 -> +8.
// Bs layout: frag = (n0/16)*(K/32) + k0/32 ; lane's 16 values contiguous at
//            Bs[frag*512 + lane*16 + i] == B[k0 + (lane>=16?8:0) + (i<8 ? i : i+8)][n0 + (lane&15)]
// Requires M % 32 == 0 (all call sites: 128 or 80000), N % 16 == 0, K % 32 == 0.
// ---------------------------------------------------------------------------
template <typename OutT>
__global__ void wmma_gemm(const unsigned short* __restrict__ A, int lda,
                          const unsigned short* __restrict__ Bs,
                          const float* __restrict__ bias,
                          OutT* __restrict__ C, int ldc,
                          int M, int N, int K) {
  int wave = threadIdx.x >> 5;
  int lane = threadIdx.x & 31;
  int m0 = blockIdx.y * 64 + (wave >> 2) * 32;
  int n0 = blockIdx.x * 128 + (wave & 3) * 32;
  if (m0 >= M || n0 >= N) return;            // wave-uniform: EXEC stays all-ones
  bool n1ok = (n0 + 16) < N;                 // second 16-col sub-tile valid?
  int hi = lane >> 4;
  int kchunks = K >> 5;
  const unsigned short* arow0  = A + (size_t)(m0 + (lane & 15)) * lda + hi * 8;
  const unsigned short* arow1  = arow0 + (size_t)16 * lda;
  const unsigned short* bfrag0 = Bs + ((size_t)(n0 >> 4) * kchunks) * 512 + (size_t)lane * 16;
  const unsigned short* bfrag1 = bfrag0 + (size_t)kchunks * 512;
  v8f acc00 = {}, acc01 = {}, acc10 = {}, acc11 = {};
  if (n1ok) {
    for (int kc = 0; kc < kchunks; ++kc) {
      int ko = kc << 5;
      v16bf a0 = wmma_cat(*reinterpret_cast<const v8u*>(arow0 + ko),
                          *reinterpret_cast<const v8u*>(arow0 + ko + 16));
      v16bf a1 = wmma_cat(*reinterpret_cast<const v8u*>(arow1 + ko),
                          *reinterpret_cast<const v8u*>(arow1 + ko + 16));
      v16bf b0 = __builtin_bit_cast(v16bf, *reinterpret_cast<const v16u*>(bfrag0 + (size_t)kc * 512));
      v16bf b1 = __builtin_bit_cast(v16bf, *reinterpret_cast<const v16u*>(bfrag1 + (size_t)kc * 512));
      acc00 = wmma_bf16(a0, b0, acc00);
      acc10 = wmma_bf16(a1, b0, acc10);
      acc01 = wmma_bf16(a0, b1, acc01);
      acc11 = wmma_bf16(a1, b1, acc11);
    }
  } else {
    for (int kc = 0; kc < kchunks; ++kc) {
      int ko = kc << 5;
      v16bf a0 = wmma_cat(*reinterpret_cast<const v8u*>(arow0 + ko),
                          *reinterpret_cast<const v8u*>(arow0 + ko + 16));
      v16bf a1 = wmma_cat(*reinterpret_cast<const v8u*>(arow1 + ko),
                          *reinterpret_cast<const v8u*>(arow1 + ko + 16));
      v16bf b0 = __builtin_bit_cast(v16bf, *reinterpret_cast<const v16u*>(bfrag0 + (size_t)kc * 512));
      acc00 = wmma_bf16(a0, b0, acc00);
      acc10 = wmma_bf16(a1, b0, acc10);
    }
  }
  int colb = n0 + (lane & 15);
  float bv0 = bias ? bias[colb] : 0.0f;
  #pragma unroll
  for (int r = 0; r < 8; ++r) {              // C layout: VGPR r -> M = r (+8 for hi lanes)
    int rr = m0 + r + hi * 8;
    store_out(&C[(size_t)rr * ldc + colb], acc00[r] + bv0);
    store_out(&C[(size_t)(rr + 16) * ldc + colb], acc10[r] + bv0);
  }
  if (n1ok) {
    float bv1 = bias ? bias[colb + 16] : 0.0f;
    #pragma unroll
    for (int r = 0; r < 8; ++r) {
      int rr = m0 + r + hi * 8;
      store_out(&C[(size_t)rr * ldc + colb + 16], acc01[r] + bv1);
      store_out(&C[(size_t)(rr + 16) * ldc + colb + 16], acc11[r] + bv1);
    }
  }
}

// ---------------------------------------------------------------------------
// Precompute helpers
// ---------------------------------------------------------------------------
__global__ void k_f32_to_bf16(const float* __restrict__ s, unsigned short* __restrict__ d,
                              long long n) {
  long long i = (long long)blockIdx.x * blockDim.x + threadIdx.x;
  long long stride = (long long)gridDim.x * blockDim.x;
  for (; i < n; i += stride) d[i] = f2bf(s[i]);
}

__global__ void k_f32_to_bf16_strided(const float* __restrict__ s, int rows, int scols,
                                      unsigned short* __restrict__ d, int dld, int dcol0) {
  int n = rows * scols;
  int i = blockIdx.x * blockDim.x + threadIdx.x;
  int stride = gridDim.x * blockDim.x;
  for (; i < n; i += stride) {
    int r = i / scols, c = i - r * scols;
    d[(size_t)r * dld + dcol0 + c] = f2bf(s[i]);
  }
}

__global__ void k_zero_bf16(unsigned short* d, long long n) {
  long long i = (long long)blockIdx.x * blockDim.x + threadIdx.x;
  long long stride = (long long)gridDim.x * blockDim.x;
  for (; i < n; i += stride) d[i] = 0;
}

// Row-major bf16 [K,N] -> WMMA B-fragment order (see wmma_gemm header comment).
__global__ void k_swizzle_b(const unsigned short* __restrict__ src,
                            unsigned short* __restrict__ dst, int K, int N) {
  int total = K * N;
  int kchunks = K >> 5;
  for (int idx = blockIdx.x * blockDim.x + threadIdx.x; idx < total;
       idx += gridDim.x * blockDim.x) {
    int f    = idx >> 9;
    int lane = (idx >> 4) & 31;
    int i    = idx & 15;
    int n0   = (f / kchunks) << 4;
    int k0   = (f % kchunks) << 5;
    int col  = n0 + (lane & 15);
    int row  = k0 + ((lane >> 4) << 3) + (i < 8 ? i : i + 8);
    dst[idx] = src[(size_t)row * N + col];
  }
}

__global__ void k_prep_bias(const float* bd, const float* bb, const float* bh0,
                            const float* bc0, const float* bih, const float* bhh,
                            const float* bfc, float* bias_dwb, float* bias_h0c0,
                            float* bias_cat, float* bias_fc) {
  int tid = threadIdx.x;
  for (int i = tid; i < 4 * H_; i += 256) bias_cat[i] = bih[i] + bhh[i];
  for (int i = tid; i < A_ + H_; i += 256) bias_dwb[i] = (i < A_) ? bd[i] : bb[i - A_];
  for (int i = tid; i < 2 * H_; i += 256) bias_h0c0[i] = (i < H_) ? bh0[i] : bc0[i - H_];
  for (int i = tid; i < 80; i += 256) bias_fc[i] = (i < V_) ? bfc[i] : 0.0f;
}

__global__ void k_mean(const float* __restrict__ enc, unsigned short* __restrict__ meanb) {
  int idx = blockIdx.x * 256 + threadIdx.x;   // 128*512
  int b = idx >> 9, e = idx & 511;
  const float* p = enc + (size_t)b * P_ * ENC_ + e;
  float s = 0.0f;
  for (int pp = 0; pp < P_; ++pp) s += p[(size_t)pp * ENC_];
  meanb[idx] = f2bf(s * (1.0f / (float)P_));
}

// h0c0[128,1024] -> c (f32), h (bf16 into xh[:,768:1280]), emb[START] into xh[:,0:256]
__global__ void k_init(const float* __restrict__ h0c0, const unsigned short* __restrict__ embb,
                       float* __restrict__ c, unsigned short* __restrict__ xh,
                       int* __restrict__ done) {
  int idx = blockIdx.x * 256 + threadIdx.x;   // 128*512
  int b = idx >> 9, j = idx & 511;
  c[idx] = h0c0[b * 1024 + 512 + j];
  xh[b * 1280 + 768 + j] = f2bf(h0c0[b * 1024 + j]);
  if (j < 256) xh[b * 1280 + j] = embb[START_TOK_ * 256 + j];
  if (idx == 0) *done = 0;
}

// ---------------------------------------------------------------------------
// Per-step kernels
// ---------------------------------------------------------------------------
// scores e[b,p] = relu(att1[b,p,:] + att2[b,:]) . wf + bf  ; then softmax -> alpha
__global__ void k_attn(const unsigned short* __restrict__ att1, const float* __restrict__ ag,
                       const float* __restrict__ wf, const float* __restrict__ bfp,
                       float* __restrict__ alpha) {
  __shared__ float att2_s[A_];
  __shared__ float wf_s[A_];
  __shared__ float e_s[640];
  __shared__ float red[256];
  int b = blockIdx.x, tid = threadIdx.x;
  att2_s[tid] = ag[b * 768 + tid];            // cols 0..255 of fused [Wd|Wb] output
  wf_s[tid]   = wf[tid];
  __syncthreads();
  int wave = tid >> 5, lane = tid & 31;
  float bfv = bfp[0];
  for (int p = wave; p < P_; p += 8) {
    v8u q = *reinterpret_cast<const v8u*>(att1 + ((size_t)(b * P_ + p)) * A_ + lane * 8);
    float s = 0.0f;
    #pragma unroll
    for (int i = 0; i < 8; ++i) {
      float v = bf2f(q[i]) + att2_s[lane * 8 + i];
      v = v > 0.0f ? v : 0.0f;
      s += v * wf_s[lane * 8 + i];
    }
    #pragma unroll
    for (int off = 16; off > 0; off >>= 1) s += __shfl_xor(s, off, 32);
    if (lane == 0) e_s[p] = s + bfv;
  }
  __syncthreads();
  float m = -3.4e38f;
  for (int p = tid; p < P_; p += 256) m = fmaxf(m, e_s[p]);
  red[tid] = m; __syncthreads();
  for (int s2 = 128; s2 > 0; s2 >>= 1) { if (tid < s2) red[tid] = fmaxf(red[tid], red[tid + s2]); __syncthreads(); }
  float mv = red[0]; __syncthreads();
  float sm = 0.0f;
  for (int p = tid; p < P_; p += 256) { float e = __expf(e_s[p] - mv); e_s[p] = e; sm += e; }
  red[tid] = sm; __syncthreads();
  for (int s2 = 128; s2 > 0; s2 >>= 1) { if (tid < s2) red[tid] += red[tid + s2]; __syncthreads(); }
  float inv = 1.0f / red[0];
  for (int p = tid; p < P_; p += 256) alpha[b * P_ + p] = e_s[p] * inv;
}

// awe[b,e] = sigmoid(gatepre[b,e]) * sum_p alpha[b,p]*enc[b,p,e]  -> xh[:,256:768] (bf16)
// 256 threads, each handles 2 adjacent columns via packed u32 loads/stores.
__global__ void k_awe(const unsigned short* __restrict__ encb, const float* __restrict__ alpha,
                      const float* __restrict__ ag, unsigned short* __restrict__ xh) {
  __shared__ float al[P_];
  int b = blockIdx.x, tid = threadIdx.x;      // 256 threads
  for (int p = tid; p < P_; p += 256) al[p] = alpha[b * P_ + p];
  __syncthreads();
  int e0 = tid * 2;
  const unsigned short* eb = encb + (size_t)b * P_ * ENC_ + e0;
  float a0 = 0.0f, a1 = 0.0f;
  for (int p = 0; p < P_; ++p) {
    unsigned int w = *reinterpret_cast<const unsigned int*>(eb + (size_t)p * ENC_);
    float ap = al[p];
    a0 += ap * bf2f((unsigned short)(w & 0xFFFFu));
    a1 += ap * bf2f((unsigned short)(w >> 16));
  }
  float g0 = sigm(ag[b * 768 + 256 + e0]);
  float g1 = sigm(ag[b * 768 + 256 + e0 + 1]);
  unsigned int wo = (unsigned int)f2bf(g0 * a0) | ((unsigned int)f2bf(g1 * a1) << 16);
  *reinterpret_cast<unsigned int*>(&xh[b * 1280 + 256 + e0]) = wo;
}

__global__ void k_lstm(const float* __restrict__ gates, float* __restrict__ c,
                       unsigned short* __restrict__ xh) {
  int idx = blockIdx.x * 256 + threadIdx.x;   // 128*512
  int b = idx >> 9, j = idx & 511;
  const float* g = gates + (size_t)b * 2048;
  float i_ = sigm(g[j]);
  float f_ = sigm(g[512 + j]);
  float gg = tanhf(g[1024 + j]);
  float o_ = sigm(g[1536 + j]);
  float cn = f_ * c[idx] + i_ * gg;
  c[idx] = cn;
  xh[b * 1280 + 768 + j] = f2bf(o_ * tanhf(cn));
}

// logits -> d_out (masked by done), flat-argmax done update, greedy emb feedback
__global__ void k_final(const float* __restrict__ preds, const unsigned short* __restrict__ embb,
                        unsigned short* __restrict__ xh, int* __restrict__ done,
                        float* __restrict__ out, int t) {
  __shared__ float rmax[128];
  __shared__ int   rarg[128];
  __shared__ int   done_s;
  int tid = threadIdx.x;
  if (tid < 128) {
    float mx = -3.4e38f; int arg = 0;
    for (int v = 0; v < V_; ++v) { float x = preds[tid * 80 + v]; if (x > mx) { mx = x; arg = v; } }
    rmax[tid] = mx; rarg[tid] = arg;
  }
  __syncthreads();
  if (tid == 0) {
    done_s = *done;
    float mx = rmax[0]; int bbst = 0;
    for (int b = 1; b < 128; ++b) if (rmax[b] > mx) { mx = rmax[b]; bbst = b; }
    int flat = bbst * V_ + rarg[bbst];
    *done = done_s | ((flat == 69) ? 1 : 0);  // END_TOK compared against FLAT argmax (ref semantics)
  }
  __syncthreads();
  int dmask = done_s;
  for (int idx = tid; idx < 128 * V_; idx += 256) {
    int b = idx / V_, v = idx - b * V_;
    out[(size_t)b * (T_ * V_) + (size_t)t * V_ + v] = dmask ? 0.0f : preds[b * 80 + v];
  }
  for (int idx = tid; idx < 128 * 256; idx += 256) {
    int b = idx >> 8, j = idx & 255;
    xh[b * 1280 + j] = embb[rarg[b] * 256 + j];
  }
}

// ---------------------------------------------------------------------------
extern "C" void kernel_launch(void* const* d_in, const int* in_sizes, int n_in,
                              void* d_out, int out_size, void* d_ws, size_t ws_size,
                              hipStream_t stream) {
  const float* enc = (const float*)d_in[0];
  const float* emb = (const float*)d_in[1];
  const float* We  = (const float*)d_in[2];
  const float* be  = (const float*)d_in[3];
  const float* Wd  = (const float*)d_in[4];
  const float* bd  = (const float*)d_in[5];
  const float* wf  = (const float*)d_in[6];
  const float* bf  = (const float*)d_in[7];
  const float* Wh0 = (const float*)d_in[8];
  const float* bh0 = (const float*)d_in[9];
  const float* Wc0 = (const float*)d_in[10];
  const float* bc0 = (const float*)d_in[11];
  const float* Wb  = (const float*)d_in[12];
  const float* bb  = (const float*)d_in[13];
  const float* Wih = (const float*)d_in[14];
  const float* Whh = (const float*)d_in[15];
  const float* bih = (const float*)d_in[16];
  const float* bhh = (const float*)d_in[17];
  const float* Wfc = (const float*)d_in[18];
  const float* bfc = (const float*)d_in[19];
  float* out = (float*)d_out;
  (void)in_sizes; (void)n_in; (void)out_size; (void)ws_size;

  char* base = (char*)d_ws;
  size_t off = 0;
  auto alloc = [&](size_t bytes) -> char* {
    char* p = base + off;
    off = (off + bytes + 255) & ~(size_t)255;
    return p;
  };
  unsigned short* enc_bf  = (unsigned short*)alloc((size_t)B_ * P_ * ENC_ * 2); // 82 MB (L2-resident)
  unsigned short* att1_bf = (unsigned short*)alloc((size_t)B_ * P_ * A_ * 2);   // 41 MB (L2-resident)
  unsigned short* We_bf   = (unsigned short*)alloc((size_t)512 * 256 * 2);
  unsigned short* WdWb_bf = (unsigned short*)alloc((size_t)512 * 768 * 2);
  unsigned short* Whc_bf  = (unsigned short*)alloc((size_t)512 * 1024 * 2);
  unsigned short* Wcat_bf = (unsigned short*)alloc((size_t)1280 * 2048 * 2);
  unsigned short* Wfc_bf  = (unsigned short*)alloc((size_t)512 * 80 * 2);
  unsigned short* We_sw   = (unsigned short*)alloc((size_t)512 * 256 * 2);      // swizzled copies
  unsigned short* WdWb_sw = (unsigned short*)alloc((size_t)512 * 768 * 2);
  unsigned short* Whc_sw  = (unsigned short*)alloc((size_t)512 * 1024 * 2);
  unsigned short* Wcat_sw = (unsigned short*)alloc((size_t)1280 * 2048 * 2);
  unsigned short* Wfc_sw  = (unsigned short*)alloc((size_t)512 * 80 * 2);
  unsigned short* emb_bf  = (unsigned short*)alloc((size_t)V_ * E_ * 2);
  unsigned short* mean_bf = (unsigned short*)alloc((size_t)B_ * ENC_ * 2);
  unsigned short* xh      = (unsigned short*)alloc((size_t)B_ * 1280 * 2);      // [emb|awe|h]
  float* bias_dwb  = (float*)alloc(768 * 4);
  float* bias_h0c0 = (float*)alloc(1024 * 4);
  float* bias_cat  = (float*)alloc(2048 * 4);
  float* bias_fc   = (float*)alloc(80 * 4);
  float* ag        = (float*)alloc((size_t)B_ * 768 * 4);
  float* alpha     = (float*)alloc((size_t)B_ * P_ * 4);
  float* gates     = (float*)alloc((size_t)B_ * 2048 * 4);
  float* cstate    = (float*)alloc((size_t)B_ * 512 * 4);
  float* h0c0      = (float*)alloc((size_t)B_ * 1024 * 4);
  float* preds     = (float*)alloc((size_t)B_ * 80 * 4);
  int*   done      = (int*)alloc(256);

  dim3 b256(256);
  // ---- one-time precompute: bf16 conversion + B-fragment swizzle ----
  k_f32_to_bf16<<<512, b256, 0, stream>>>(We, We_bf, 512LL * 256);
  k_f32_to_bf16<<<512, b256, 0, stream>>>(emb, emb_bf, (long long)V_ * E_);
  k_f32_to_bf16<<<2048, b256, 0, stream>>>(Wih, Wcat_bf, 768LL * 2048);
  k_f32_to_bf16<<<2048, b256, 0, stream>>>(Whh, Wcat_bf + (size_t)768 * 2048, 512LL * 2048);
  k_f32_to_bf16_strided<<<512, b256, 0, stream>>>(Wd, 512, 256, WdWb_bf, 768, 0);
  k_f32_to_bf16_strided<<<512, b256, 0, stream>>>(Wb, 512, 512, WdWb_bf, 768, 256);
  k_f32_to_bf16_strided<<<512, b256, 0, stream>>>(Wh0, 512, 512, Whc_bf, 1024, 0);
  k_f32_to_bf16_strided<<<512, b256, 0, stream>>>(Wc0, 512, 512, Whc_bf, 1024, 512);
  k_zero_bf16<<<160, b256, 0, stream>>>(Wfc_bf, 512LL * 80);
  k_f32_to_bf16_strided<<<160, b256, 0, stream>>>(Wfc, 512, 70, Wfc_bf, 80, 0);
  k_swizzle_b<<<512,  b256, 0, stream>>>(We_bf,   We_sw,   512, 256);
  k_swizzle_b<<<1024, b256, 0, stream>>>(WdWb_bf, WdWb_sw, 512, 768);
  k_swizzle_b<<<1024, b256, 0, stream>>>(Whc_bf,  Whc_sw,  512, 1024);
  k_swizzle_b<<<4096, b256, 0, stream>>>(Wcat_bf, Wcat_sw, 1280, 2048);
  k_swizzle_b<<<160,  b256, 0, stream>>>(Wfc_bf,  Wfc_sw,  512, 80);
  k_prep_bias<<<1, b256, 0, stream>>>(bd, bb, bh0, bc0, bih, bhh, bfc,
                                      bias_dwb, bias_h0c0, bias_cat, bias_fc);
  k_f32_to_bf16<<<4096, b256, 0, stream>>>(enc, enc_bf, (long long)B_ * P_ * ENC_);
  k_mean<<<256, b256, 0, stream>>>(enc, mean_bf);

  // att1 = enc @ We + be  (bf16 output, 21 GFLOP via WMMA)
  wmma_gemm<unsigned short><<<dim3(2, (B_ * P_) / 64), b256, 0, stream>>>(
      enc_bf, 512, We_sw, be, att1_bf, 256, B_ * P_, 256, 512);
  // [h0|c0] = mean @ [Wh0|Wc0] + [bh0|bc0]
  wmma_gemm<float><<<dim3(8, 2), b256, 0, stream>>>(
      mean_bf, 512, Whc_sw, bias_h0c0, h0c0, 1024, B_, 1024, 512);
  k_init<<<256, b256, 0, stream>>>(h0c0, emb_bf, cstate, xh, done);

  // ---- sequential decode: 70 steps ----
  for (int t = 0; t < T_; ++t) {
    // [att2 | gate_pre] = h @ [Wd|Wb] + [bd|bb]
    wmma_gemm<float><<<dim3(6, 2), b256, 0, stream>>>(
        xh + 768, 1280, WdWb_sw, bias_dwb, ag, 768, B_, 768, 512);
    k_attn<<<B_, b256, 0, stream>>>(att1_bf, ag, wf, bf, alpha);
    k_awe<<<B_, b256, 0, stream>>>(enc_bf, alpha, ag, xh);
    // gates = [emb|awe|h] @ [Wih;Whh] + (bih+bhh)
    wmma_gemm<float><<<dim3(16, 2), b256, 0, stream>>>(
        xh, 1280, Wcat_sw, bias_cat, gates, 2048, B_, 2048, 1280);
    k_lstm<<<256, b256, 0, stream>>>(gates, cstate, xh);
    // preds = h_new @ Wfc(padded to 80) + bfc
    wmma_gemm<float><<<dim3(1, 2), b256, 0, stream>>>(
        xh + 768, 1280, Wfc_sw, bias_fc, preds, 80, B_, 80, 512);
    k_final<<<1, b256, 0, stream>>>(preds, emb_bf, xh, done, out, t);
  }
}
